// BERTCharting_60971355734708
// MI455X (gfx1250) — compile-verified, matrix operand order checked by reading, not run on hardware
//
#include <hip/hip_runtime.h>
#include <hip/hip_fp16.h>

typedef __attribute__((ext_vector_type(16))) _Float16 v16h;
typedef __attribute__((ext_vector_type(8)))  _Float16 v8h;
typedef __attribute__((ext_vector_type(4)))  _Float16 v4h;
typedef __attribute__((ext_vector_type(8)))  float    v8f;
typedef __attribute__((ext_vector_type(4)))  float    v4f;

// Problem sizes
constexpr int Bb   = 8;
constexpr int Nn   = 128;
constexpr int Hh   = 768;
constexpr int HID  = 384;
constexpr int Ll   = 100;
constexpr int M1   = Bb * Nn;     // 1024 rows of X
constexpr int PQW  = 2 * HID;     // 768 (p | q)
constexpr int KT1  = Hh / 32;     // 24 K-tiles for GEMM1
constexpr int NT1  = PQW / 16;    // 48 N-tiles for GEMM1
constexpr int KT2  = HID / 32;    // 12 K-tiles for GEMM2
constexpr int NT2  = 7;           // ceil(100/16) N-tiles for GEMM2

// ---------------------------------------------------------------------------
// Convert X (B*N x H) f32 -> f16 row-major (A-operand of GEMM1)
// ---------------------------------------------------------------------------
__global__ void cvt_x_kernel(const float* __restrict__ x, _Float16* __restrict__ xh) {
  int t = blockIdx.x * blockDim.x + threadIdx.x;   // one float4 per thread
  v4f v = ((const v4f*)x)[t];
  v4h o;
  o[0] = (_Float16)v[0]; o[1] = (_Float16)v[1];
  o[2] = (_Float16)v[2]; o[3] = (_Float16)v[3];
  ((v4h*)xh)[t] = o;
}

// ---------------------------------------------------------------------------
// Pack [W1_row | W1_col] (768 x 768 logical, f32) into per-lane WMMA
// B-fragment layout: frag[kt][nt][lane][16 halves].
// 16-bit B 32x16: lane<16 -> N=lane,    halves 0..7 = K k0+0..7,  8..15 = K k0+16..23
//                 lane>=16-> N=lane-16, halves 0..7 = K k0+8..15, 8..15 = K k0+24..31
// ---------------------------------------------------------------------------
__global__ void pack_w1_kernel(const float* __restrict__ W1, _Float16* __restrict__ frag) {
  int t = blockIdx.x * blockDim.x + threadIdx.x;
  if (t >= KT1 * NT1 * 32) return;
  int lane = t & 31;
  int nt   = (t >> 5) % NT1;
  int kt   = (t >> 5) / NT1;
  int half = lane >> 4;
  int n    = nt * 16 + (lane & 15);
  v16h o;
  for (int oidx = 0; oidx < 16; ++oidx) {
    int k = kt * 32 + ((oidx < 8) ? (8 * half + oidx) : (16 + 8 * half + (oidx - 8)));
    float f = (n < HID) ? W1[k * HID + n] : W1[(k + Hh) * HID + (n - HID)];
    o[oidx] = (_Float16)f;
  }
  ((v16h*)frag)[t] = o;
}

// ---------------------------------------------------------------------------
// Pack W2 (384 x 100, f32) into B-fragment layout, N zero-padded to 112.
// ---------------------------------------------------------------------------
__global__ void pack_w2_kernel(const float* __restrict__ W2, _Float16* __restrict__ frag) {
  int t = blockIdx.x * blockDim.x + threadIdx.x;
  if (t >= KT2 * NT2 * 32) return;
  int lane = t & 31;
  int nt   = (t >> 5) % NT2;
  int kt   = (t >> 5) / NT2;
  int half = lane >> 4;
  int n    = nt * 16 + (lane & 15);
  v16h o;
  for (int oidx = 0; oidx < 16; ++oidx) {
    int k = kt * 32 + ((oidx < 8) ? (8 * half + oidx) : (16 + 8 * half + (oidx - 8)));
    float f = (n < Ll) ? W2[k * Ll + n] : 0.0f;
    o[oidx] = (_Float16)f;
  }
  ((v16h*)frag)[t] = o;
}

// ---------------------------------------------------------------------------
// GEMM1: PQ (1024 x 768, f32) = Xh (1024x768 f16) @ W1full (768x768 f16)
// One wave per workgroup. Wave owns one 16-row M-tile and 4 N-tiles (16x64).
// ---------------------------------------------------------------------------
__global__ void __launch_bounds__(32) gemm1_kernel(const _Float16* __restrict__ Xh,
                                                   const _Float16* __restrict__ W1f,
                                                   float* __restrict__ PQ) {
  const int mt   = blockIdx.x;           // 0..63
  const int ntb  = blockIdx.y * 4;       // 0,4,...,44
  const int lane = threadIdx.x;
  const int half = lane >> 4;
  const int mrow = (mt << 4) + (lane & 15);
  const _Float16* arow = Xh + (size_t)mrow * PQW;

  v8f acc[4] = {};
  for (int kt = 0; kt < KT1; ++kt) {
    const int k0 = kt * 32;
    v8h a0 = *(const v8h*)(arow + k0 + 8 * half);
    v8h a1 = *(const v8h*)(arow + k0 + 16 + 8 * half);
    v16h a;
    for (int o = 0; o < 8; ++o) { a[o] = a0[o]; a[o + 8] = a1[o]; }

    const v16h* bf = (const v16h*)W1f + (size_t)(kt * NT1 + ntb) * 32 + lane;
    for (int tle = 0; tle < 4; ++tle) {
      v16h b = bf[tle * 32];
      acc[tle] = __builtin_amdgcn_wmma_f32_16x16x32_f16(
          false, a, false, b, (short)0, acc[tle], false, false);
    }
  }
  for (int tle = 0; tle < 4; ++tle) {
    int n = ((ntb + tle) << 4) + (lane & 15);
    for (int r = 0; r < 8; ++r) {
      int m = (mt << 4) + r + 8 * half;
      PQ[(size_t)m * PQW + n] = acc[tle][r];
    }
  }
}

// ---------------------------------------------------------------------------
// GEMM2 (fused bilinear head): one workgroup per (b,i).
//  - W2 fragment image (84 KB) staged LDS-resident once per WG via
//    global_load_async_to_lds_b128 (ASYNCcnt async-DMA path).
//  - qb[d] = q[b,i,d] + b1[d] staged once into LDS.
//  - Each of 8 waves owns one 16-row j-tile; builds A = relu(p + qb) in f16
//    on the fly; 12 K-steps x 7 N-tiles of v_wmma_f32_16x16x32_f16 with B
//    operands from ds_load_b128; adds b2 and stores f32.
// ---------------------------------------------------------------------------
struct Gemm2Smem {
  __align__(16) _Float16 w2[KT2 * NT2 * 32 * 16];  // 86,016 B fragment image
  __align__(16) float    qb[HID];                  //  1,536 B
};

__global__ void __launch_bounds__(256) gemm2_kernel(const float* __restrict__ PQ,
                                                    const float* __restrict__ b1,
                                                    const _Float16* __restrict__ W2f,
                                                    const float* __restrict__ b2,
                                                    float* __restrict__ out) {
  __shared__ Gemm2Smem smem;

  const int b = blockIdx.x >> 7;
  const int i = blockIdx.x & 127;

  // --- Async-stage the whole W2 fragment image into LDS (5376 x b128) ---
  {
    const unsigned int lds_base = (unsigned int)(uintptr_t)(&smem.w2[0]);
    const char* gsrc = (const char*)W2f;
    constexpr int N128 = (int)(sizeof(smem.w2) / 16);   // 5376
    for (int t = threadIdx.x; t < N128; t += 256) {
      unsigned int       ldsa = lds_base + (unsigned int)(t * 16);
      unsigned long long ga   = (unsigned long long)(uintptr_t)(gsrc + (size_t)t * 16);
      asm volatile("global_load_async_to_lds_b128 %0, %1, off"
                   :: "v"(ldsa), "v"(ga) : "memory");
    }
  }

  // --- qb = q[b,i,:] + b1 (regular path; needs arithmetic) ---
  const float* qrow = PQ + (size_t)(b * Nn + i) * PQW + HID;
  for (int d = threadIdx.x; d < HID; d += blockDim.x)
    smem.qb[d] = qrow[d] + b1[d];

  asm volatile("s_wait_asynccnt 0x0" ::: "memory");
  __syncthreads();

  const int wave = threadIdx.x >> 5;
  const int lane = threadIdx.x & 31;
  const int half = lane >> 4;
  const int j    = (wave << 4) + (lane & 15);
  const float* prow = PQ + (size_t)(b * Nn + j) * PQW;
  const v16h* bfl = (const v16h*)(&smem.w2[0]);

  v8f acc[NT2] = {};
  for (int kt = 0; kt < KT2; ++kt) {
    const int k0 = kt * 32;
    const int ka = k0 + 8 * half;        // halves 0..7
    const int kb = k0 + 16 + 8 * half;   // halves 8..15
    v4f pa0 = *(const v4f*)(prow + ka);
    v4f pa1 = *(const v4f*)(prow + ka + 4);
    v4f pb0 = *(const v4f*)(prow + kb);
    v4f pb1 = *(const v4f*)(prow + kb + 4);
    v4f qa0 = *(const v4f*)(smem.qb + ka);
    v4f qa1 = *(const v4f*)(smem.qb + ka + 4);
    v4f qb0 = *(const v4f*)(smem.qb + kb);
    v4f qb1 = *(const v4f*)(smem.qb + kb + 4);

    v16h a;
    for (int o = 0; o < 4; ++o) {
      float h0 = pa0[o] + qa0[o]; h0 = h0 > 0.0f ? h0 : 0.0f;
      float h1 = pa1[o] + qa1[o]; h1 = h1 > 0.0f ? h1 : 0.0f;
      float h2 = pb0[o] + qb0[o]; h2 = h2 > 0.0f ? h2 : 0.0f;
      float h3 = pb1[o] + qb1[o]; h3 = h3 > 0.0f ? h3 : 0.0f;
      a[o]      = (_Float16)h0;
      a[o + 4]  = (_Float16)h1;
      a[o + 8]  = (_Float16)h2;
      a[o + 12] = (_Float16)h3;
    }

    for (int tle = 0; tle < NT2; ++tle) {
      v16h bb = bfl[(kt * NT2 + tle) * 32 + lane];   // ds_load_b128 x2
      acc[tle] = __builtin_amdgcn_wmma_f32_16x16x32_f16(
          false, a, false, bb, (short)0, acc[tle], false, false);
    }
  }

  // Epilogue: out[b, i, j, l] = acc + b2[l], drop padded l >= 100 columns
  float* obase = out + (size_t)(b * Nn + i) * Nn * Ll;
  for (int tle = 0; tle < NT2; ++tle) {
    int n = (tle << 4) + (lane & 15);
    if (n < Ll) {
      float bias = b2[n];
      for (int r = 0; r < 8; ++r) {
        int jj = (wave << 4) + r + 8 * half;
        obase[(size_t)jj * Ll + n] = acc[tle][r] + bias;
      }
    }
  }
}

// ---------------------------------------------------------------------------
extern "C" void kernel_launch(void* const* d_in, const int* in_sizes, int n_in,
                              void* d_out, int out_size, void* d_ws, size_t ws_size,
                              hipStream_t stream) {
  const float* repr = (const float*)d_in[0];   // (8,128,768)
  const float* W1   = (const float*)d_in[1];   // (1536,384)
  const float* b1   = (const float*)d_in[2];   // (384,)
  const float* W2   = (const float*)d_in[3];   // (384,100)
  const float* b2   = (const float*)d_in[4];   // (100,)
  float* out = (float*)d_out;                  // (8,128,128,100)

  // Workspace layout (all offsets 256B aligned)
  char* ws = (char*)d_ws;
  _Float16* Xh  = (_Float16*)(ws);                 // 1024*768*2  = 1,572,864 B
  _Float16* W1f = (_Float16*)(ws + 1572864);       // 24*48*32*32 = 1,179,648 B
  _Float16* W2f = (_Float16*)(ws + 2752512);       // 12*7*32*32  =    86,016 B
  float*    PQ  = (float*)   (ws + 2838528);       // 1024*768*4  = 3,145,728 B

  // 1) X -> f16 (786432 elems, 4 per thread)
  cvt_x_kernel<<<768, 256, 0, stream>>>(repr, Xh);

  // 2) Pack weights into WMMA B-fragment layout
  pack_w1_kernel<<<(KT1 * NT1 * 32 + 255) / 256, 256, 0, stream>>>(W1, W1f);
  pack_w2_kernel<<<(KT2 * NT2 * 32 + 255) / 256, 256, 0, stream>>>(W2, W2f);

  // 3) PQ = Xh @ [W1_row | W1_col]   (f32 accum)
  dim3 g1(M1 / 16, NT1 / 4);
  gemm1_kernel<<<g1, 32, 0, stream>>>(Xh, W1f, PQ);

  // 4) Fused relu-bilinear head, one WG per (b,i)
  gemm2_kernel<<<Bb * Nn, 256, 0, stream>>>(PQ, b1, W2f, b2, out);
}
